// RelationalGraphConvolution_3032246911556
// MI455X (gfx1250) — matrix operand order repaired and checked by reading, not compiled, not run on hardware
//
#include <hip/hip_runtime.h>
#include <hip/hip_bf16.h>
#include <math.h>

typedef __bf16 bf16;
typedef __attribute__((ext_vector_type(16))) __bf16 v16bf;
typedef __attribute__((ext_vector_type(8)))  __bf16 v8bf;
typedef __attribute__((ext_vector_type(8)))  float  v8f;
typedef __attribute__((ext_vector_type(4)))  float  f32x4;

// Problem constants (match the reference)
constexpr int B_  = 4;
constexpr int N_  = 1024;
constexpr int R_  = 16;
constexpr int DE  = 128;
constexpr int DR  = 64;
constexpr int NB  = 192;                 // num_bases
constexpr int OD  = 128;                 // out_dim
constexpr int KE  = 144;                 // extended node feature dim: 128 + ones col + 15 zero pad
constexpr int SUP = R_ * (DE + DR);      // 3072 supports feature dim

// ---------------------------------------------------------------------------
// Prep: nodeT[b, d, j] = (d<128) ? node[b,j,d] : (d==128 ? 1 : 0), bf16
// Transposed so stage-1 WMMA B-matrix columns are contiguous in memory.
// ---------------------------------------------------------------------------
__global__ void k_prep_node(const float* __restrict__ node, bf16* __restrict__ nodeT) {
    int idx = blockIdx.x * blockDim.x + threadIdx.x;      // B*KE*N threads
    int j = idx % N_;
    int d = (idx / N_) % KE;
    int b = idx / (N_ * KE);
    float v;
    if (d < DE)       v = node[((size_t)(b * N_ + j)) * DE + d];
    else if (d == DE) v = 1.0f;                            // ones column -> row_sum
    else              v = 0.0f;                            // pad
    nodeT[idx] = (bf16)v;
}

// Prep: Wb f32 -> bf16 (same row-major 192x3072 layout)
__global__ void k_prep_wb(const float* __restrict__ wb, bf16* __restrict__ wbh) {
    int idx = blockIdx.x * blockDim.x + threadIdx.x;      // NB*SUP threads
    wbh[idx] = (bf16)wb[idx];
}

// ---------------------------------------------------------------------------
// Stage 1: per (b, r, 16-row tile): C(16x144) = adj_tile(16x1024) x nodeT(1024x144)
// bf16 WMMA, f32 accumulate. Writes node part of `supports` + row_sum.
// ---------------------------------------------------------------------------
__global__ void __launch_bounds__(256)
k_stage1(const float* __restrict__ adj, const bf16* __restrict__ nodeT,
         bf16* __restrict__ supports, float* __restrict__ rowsum) {
    const int lane  = threadIdx.x & 31;
    const int wid   = blockIdx.x * (blockDim.x >> 5) + (threadIdx.x >> 5);
    const int itile = wid & 63;              // 64 row tiles
    const int r     = (wid >> 6) & 15;       // 16 relations
    const int b     = wid >> 10;             // 4 batches

    const int mrow = lane & 15;              // A row / B,C column selector
    const int hsel = lane >> 4;              // half-wave select

    const float* arow = adj + ((size_t)((b * R_ + r) * N_) + (size_t)itile * 16 + mrow) * N_;

    v8f acc[9];
#pragma unroll
    for (int t = 0; t < 9; ++t) acc[t] = v8f{};

    for (int kk = 0; kk < N_; kk += 32) {
        // ---- A tile: 16x32 f32 -> bf16. Lane L: row M=L&15, K chunks {ka, ka+16}
        const int ka = kk + hsel * 8;
        f32x4 a0 = *(const f32x4*)(arow + ka);
        f32x4 a1 = *(const f32x4*)(arow + ka + 4);
        f32x4 a2 = *(const f32x4*)(arow + ka + 16);
        f32x4 a3 = *(const f32x4*)(arow + ka + 20);
        if (kk + 128 < N_) __builtin_prefetch(arow + kk + 128, 0, 0);
        v16bf av;
#pragma unroll
        for (int e = 0; e < 4; ++e) {
            av[e]      = (bf16)a0[e];
            av[4 + e]  = (bf16)a1[e];
            av[8 + e]  = (bf16)a2[e];
            av[12 + e] = (bf16)a3[e];
        }
        const int kb = kk + hsel * 8;        // B: lane holds K chunks {kb, kb+16}
#pragma unroll
        for (int t = 0; t < 9; ++t) {
            const int n = t * 16 + mrow;     // extended feature column 0..143
            const bf16* bp = nodeT + (size_t)(b * KE + n) * N_ + kb;
            v8bf blo = *(const v8bf*)(bp);
            v8bf bhi = *(const v8bf*)(bp + 16);
            v16bf bv;
#pragma unroll
            for (int e = 0; e < 8; ++e) { bv[e] = blo[e]; bv[8 + e] = bhi[e]; }
            acc[t] = __builtin_amdgcn_wmma_f32_16x16x32_bf16(
                false, av, false, bv, (short)0, acc[t], false, false);
        }
    }

    // ---- store node part into supports (torch concat order: col = r*192 + d)
    const int rowbase = itile * 16 + hsel * 8;
#pragma unroll
    for (int t = 0; t < 8; ++t) {
        const int d = t * 16 + mrow;
#pragma unroll
        for (int v = 0; v < 8; ++v) {
            const int i = rowbase + v;
            supports[(size_t)(b * N_ + i) * SUP + r * (DE + DR) + d] = (bf16)acc[t][v];
        }
    }
    // ---- row_sum = extended column d==128 (ones column)
    if (mrow == 0) {
#pragma unroll
        for (int v = 0; v < 8; ++v)
            rowsum[(size_t)((b * R_ + r) * N_) + rowbase + v] = acc[8][v];
    }
}

// ---------------------------------------------------------------------------
// Fill relation part: supports[b, n, r*192 + 128 + c] = row_sum[b,r,n] * rel[b,r,c]
// ---------------------------------------------------------------------------
__global__ void k_relfill(const float* __restrict__ rowsum, const float* __restrict__ rel,
                          bf16* __restrict__ supports) {
    int idx = blockIdx.x * blockDim.x + threadIdx.x;      // B*N*R*DR threads
    int c = idx & 63;
    int r = (idx >> 6) & 15;
    int n = (idx >> 10) & 1023;
    int b = idx >> 20;
    float rs = rowsum[((b * R_ + r) * N_) + n];
    float rv = rel[(b * R_ + r) * DR + c];
    supports[(size_t)(b * N_ + n) * SUP + r * (DE + DR) + DE + c] = (bf16)(rs * rv);
}

// ---------------------------------------------------------------------------
// Stage 2: h(4096x192) = supports(4096x3072 bf16) @ Wb.T  (B cols = Wb rows, contiguous)
// ---------------------------------------------------------------------------
__global__ void __launch_bounds__(256)
k_stage2(const bf16* __restrict__ supports, const bf16* __restrict__ wbh,
         float* __restrict__ h) {
    const int lane  = threadIdx.x & 31;
    const int wid   = blockIdx.x * (blockDim.x >> 5) + (threadIdx.x >> 5);
    const int ntile = wid % (NB / 16);       // 12 base tiles
    const int mtile = wid / (NB / 16);       // 256 row tiles
    const int mrow  = lane & 15;
    const int hsel  = lane >> 4;

    const bf16* arow = supports + (size_t)(mtile * 16 + mrow) * SUP;
    const bf16* brow = wbh      + (size_t)(ntile * 16 + mrow) * SUP;

    v8f acc = v8f{};
    for (int kk = 0; kk < SUP; kk += 32) {
        const int ka = kk + hsel * 8;
        v8bf alo = *(const v8bf*)(arow + ka);
        v8bf ahi = *(const v8bf*)(arow + ka + 16);
        v16bf av;
#pragma unroll
        for (int e = 0; e < 8; ++e) { av[e] = alo[e]; av[8 + e] = ahi[e]; }
        const int kb = kk + hsel * 8;
        v8bf blo = *(const v8bf*)(brow + kb);
        v8bf bhi = *(const v8bf*)(brow + kb + 16);
        v16bf bv;
#pragma unroll
        for (int e = 0; e < 8; ++e) { bv[e] = blo[e]; bv[8 + e] = bhi[e]; }
        acc = __builtin_amdgcn_wmma_f32_16x16x32_bf16(
            false, av, false, bv, (short)0, acc, false, false);
    }
#pragma unroll
    for (int v = 0; v < 8; ++v) {
        const int m = mtile * 16 + hsel * 8 + v;
        h[(size_t)m * NB + ntile * 16 + mrow] = acc[v];
    }
}

// ---------------------------------------------------------------------------
// Stage 3: out = sigmoid(h @ Ww.T + bias), f32 VALU (tiny: 0.2 GFLOP)
// ---------------------------------------------------------------------------
__global__ void k_stage3(const float* __restrict__ h, const float* __restrict__ ww,
                         const float* __restrict__ bias, float* __restrict__ out) {
    int idx = blockIdx.x * blockDim.x + threadIdx.x;      // (B*N)*OD threads
    int o = idx & (OD - 1);
    int m = idx >> 7;
    const float* hr = h  + (size_t)m * NB;
    const float* wr = ww + (size_t)o * NB;
    float acc = bias[o];
#pragma unroll 4
    for (int k = 0; k < NB; ++k) acc = fmaf(hr[k], wr[k], acc);
    out[idx] = 1.0f / (1.0f + __expf(-acc));
}

// ---------------------------------------------------------------------------
extern "C" void kernel_launch(void* const* d_in, const int* in_sizes, int n_in,
                              void* d_out, int out_size, void* d_ws, size_t ws_size,
                              hipStream_t stream) {
    const float* node = (const float*)d_in[0];   // (4,1024,128)
    const float* rel  = (const float*)d_in[1];   // (4,16,64)
    const float* adj  = (const float*)d_in[2];   // (4,16,1024,1024)
    const float* Wb   = (const float*)d_in[3];   // (192,3072)
    const float* Ww   = (const float*)d_in[4];   // (128,192)
    const float* bias = (const float*)d_in[5];   // (128,)
    float* out = (float*)d_out;                  // (4,1024,128)

    char* ws = (char*)d_ws;
    bf16*  nodeT  = (bf16*)(ws);                 //  1,179,648 B  (B*KE*N bf16)
    bf16*  wbh    = (bf16*)(ws +  1179648);      //  1,179,648 B  (NB*SUP bf16)
    bf16*  sup    = (bf16*)(ws +  2359296);      // 25,165,824 B  (B*N*SUP bf16)
    float* rowsum = (float*)(ws + 27525120);     //    262,144 B  (B*R*N f32)
    float* hbuf   = (float*)(ws + 27787264);     //  3,145,728 B  (B*N*NB f32)
    // total ws use: ~30.9 MB

    k_prep_node<<<(B_ * KE * N_) / 256, 256, 0, stream>>>(node, nodeT);
    k_prep_wb  <<<(NB * SUP) / 256,     256, 0, stream>>>(Wb, wbh);
    k_stage1   <<<(B_ * R_ * (N_ / 16)) / 8, 256, 0, stream>>>(adj, nodeT, sup, rowsum);
    k_relfill  <<<(B_ * N_ * R_ * DR) / 256, 256, 0, stream>>>(rowsum, rel, sup);
    k_stage2   <<<((B_ * N_ / 16) * (NB / 16)) / 8, 256, 0, stream>>>(sup, wbh, hbuf);
    k_stage3   <<<(B_ * N_ * OD) / 256, 256, 0, stream>>>(hbuf, Ww, bias, out);
}
